// CrossAttention3D_77300821393993
// MI455X (gfx1250) — compile-verified, matrix operand order checked by reading, not run on hardware
//
#include <hip/hip_runtime.h>
#include <hip/hip_bf16.h>
#include <math.h>

// ---------------------------------------------------------------------------
// CDNA5 (gfx1250) cross-attention: wave32, v_wmma_f32_16x16x32_f16 matmuls,
// async global->LDS staging (ASYNCcnt) for the V operand.
// ---------------------------------------------------------------------------

typedef __attribute__((ext_vector_type(16))) _Float16 v16h;
typedef __attribute__((ext_vector_type(8)))  float    v8f;

static constexpr int kDim = 256;   // model dim (single head, softmax scale 1/16)
static constexpr int kKC  = 64;    // k-chunk per flash-attention iteration
static constexpr int kL   = 4096;  // sequence length (Lq == Lk)

// WMMA 16x16x32 f16 fragment loaders (wave32 layouts per CDNA5 ISA 7.12.2).
// A-matrix 16x32: lane<16 -> row=lane, K = {kh*8..+7, 16+kh*8..+7}, kh=lane>>4.
// A row-major A-fragment of K is exactly the B-fragment of K^T (layout duality),
// so both A and B fragments load as two contiguous 16-byte chunks per lane.
__device__ inline v16h load_a_frag_f16(const _Float16* base, int ld, int lane) {
  const int m  = lane & 15;
  const int kh = lane >> 4;
  const _Float16* p = base + m * ld + kh * 8;
  v16h a;
#pragma unroll
  for (int j = 0; j < 8; ++j) {
    a[j]     = p[j];        // K = kh*8 + j
    a[j + 8] = p[16 + j];   // K = 16 + kh*8 + j
  }
  return a;
}

__device__ inline v16h load_a_frag_f32(const float* base, int ld, int lane) {
  const int m  = lane & 15;
  const int kh = lane >> 4;
  const float* p = base + m * ld + kh * 8;
  v16h a;
#pragma unroll
  for (int j = 0; j < 8; ++j) {
    a[j]     = (_Float16)p[j];
    a[j + 8] = (_Float16)p[16 + j];
  }
  return a;
}

__device__ inline v8f wmma_f16(v16h a, v16h b, v8f c) {
  return __builtin_amdgcn_wmma_f32_16x16x32_f16(false, a, false, b, (short)0, c,
                                                false, false);
}

// ---------------------------------------------------------------------------
// Projection: Y[N][256] = f16( X[N][256] @ W[256][256] + bias )
// Block: 128 threads = 4 waves; 16 rows x 64 cols; wave w owns cols w*16..+15.
// ---------------------------------------------------------------------------
__global__ __launch_bounds__(128) void proj_kernel(const float* __restrict__ X,
                                                   const float* __restrict__ W,
                                                   const float* __restrict__ bias,
                                                   _Float16* __restrict__ Y) {
  __shared__ _Float16 Wt[64][32];  // transposed W chunk: Wt[n][kk]

  const int tid  = threadIdx.x;
  const int lane = tid & 31;
  const int w    = tid >> 5;
  const int rbase = blockIdx.x * 16;
  const int cbase = blockIdx.y * 64;

  v8f acc = {};
  for (int kc = 0; kc < kDim; kc += 32) {
    for (int e = tid; e < 64 * 32; e += 128) {
      const int kk = e >> 6;
      const int n  = e & 63;
      Wt[n][kk] = (_Float16)W[(size_t)(kc + kk) * kDim + cbase + n];
    }
    __syncthreads();

    v16h a = load_a_frag_f32(X + (size_t)rbase * kDim + kc, kDim, lane);
    v16h b = load_a_frag_f16(&Wt[w * 16][0], 32, lane);   // B-frag of W chunk
    acc = wmma_f16(a, b, acc);
    __syncthreads();
  }

  const int col = cbase + w * 16 + (lane & 15);
  const float bv = bias[col];
#pragma unroll
  for (int r = 0; r < 8; ++r) {
    const int row = rbase + r + 8 * (lane >> 4);
    Y[(size_t)row * kDim + col] = (_Float16)(acc[r] + bv);
  }
}

// ---------------------------------------------------------------------------
// Projection with TRANSPOSED output (for V): YT[b][col][l] = f16(X@W + b).
// Rows r=0..7 of the C tile are consecutive along l in the transposed layout,
// so each lane emits a single contiguous 16-byte store.
// ---------------------------------------------------------------------------
__global__ __launch_bounds__(128) void projT_kernel(const float* __restrict__ X,
                                                    const float* __restrict__ W,
                                                    const float* __restrict__ bias,
                                                    _Float16* __restrict__ YT) {
  __shared__ _Float16 Wt[64][32];

  const int tid  = threadIdx.x;
  const int lane = tid & 31;
  const int w    = tid >> 5;
  const int rbase = blockIdx.x * 16;
  const int cbase = blockIdx.y * 64;

  v8f acc = {};
  for (int kc = 0; kc < kDim; kc += 32) {
    for (int e = tid; e < 64 * 32; e += 128) {
      const int kk = e >> 6;
      const int n  = e & 63;
      Wt[n][kk] = (_Float16)W[(size_t)(kc + kk) * kDim + cbase + n];
    }
    __syncthreads();

    v16h a = load_a_frag_f32(X + (size_t)rbase * kDim + kc, kDim, lane);
    v16h b = load_a_frag_f16(&Wt[w * 16][0], 32, lane);
    acc = wmma_f16(a, b, acc);
    __syncthreads();
  }

  const int col  = cbase + w * 16 + (lane & 15);
  const float bv = bias[col];
  const int row0 = rbase + 8 * (lane >> 4);   // 8 consecutive rows per lane
  const int b    = row0 >> 12;                // 4096 rows per batch
  const int l0   = row0 & (kL - 1);

  alignas(16) _Float16 tmp[8];
#pragma unroll
  for (int r = 0; r < 8; ++r) tmp[r] = (_Float16)(acc[r] + bv);
  *(uint4*)&YT[((size_t)b * kDim + col) * kL + l0] = *(const uint4*)tmp;
}

// ---------------------------------------------------------------------------
// Flash attention. Block = 256 threads = 8 waves, 32 q-rows per block.
// Wave w: qh = w>>2 selects q half (16 rows), kt = w&3 selects both the S
// k-tile (16 cols of the 64-wide chunk) and the O d-slice (64 cols of 256).
// Per chunk: 8 S-WMMAs + 8 PV-WMMAs per wave (balanced).
// V arrives pre-transposed ([B][D][Lk]); the per-chunk stage is a contiguous
// 32 KB copy issued as global_load_async_to_lds_b128 (ASYNCcnt-tracked).
// ---------------------------------------------------------------------------
__global__ __launch_bounds__(256) void attn_kernel(const _Float16* __restrict__ Qh,
                                                   const _Float16* __restrict__ Kh,
                                                   const _Float16* __restrict__ VhT,
                                                   _Float16* __restrict__ Xa,
                                                   int Lq, int Lk) {
  __shared__ _Float16 Vt[kDim][kKC];   // transposed V chunk  (32 KB)
  __shared__ float    Ssc[32][kKC];    // raw scores           (8 KB)
  __shared__ _Float16 Ps[32][kKC];     // probabilities        (4 KB)
  __shared__ float    red[32][8];      // row reduction scratch
  __shared__ float    mrow[32], lrow[32], arow[32];

  const int tid  = threadIdx.x;
  const int lane = tid & 31;
  const int w    = tid >> 5;
  const int qh   = w >> 2;
  const int kt   = w & 3;
  const int b     = blockIdx.y;
  const int qbase = blockIdx.x * 32;
  const float scale = 0.0625f;  // 1/sqrt(256)

  const _Float16* Qb  = Qh  + (size_t)b * Lq * kDim;
  const _Float16* Kb  = Kh  + (size_t)b * Lk * kDim;
  const _Float16* VbT = VhT + (size_t)b * kDim * Lk;

  // Preload this wave's Q tile (16 x 256) as 8 A-fragments (reused Lk/64 times).
  v16h qfrag[8];
#pragma unroll
  for (int d = 0; d < 8; ++d)
    qfrag[d] = load_a_frag_f16(Qb + (size_t)(qbase + qh * 16) * kDim + d * 32,
                               kDim, lane);

  v8f oacc[4] = {};  // 16 q-rows x 64 d-cols slice of O, fp32

  if (tid < 32) { mrow[tid] = -3.0e38f; lrow[tid] = 0.0f; }
  __syncthreads();

  for (int kb = 0; kb < Lk; kb += kKC) {
    // ---- stage V^T chunk: Vt[d][0..63] = VbT[d][kb..kb+63]; pure contiguous
    // copy, 16 B per lane per op, async global->LDS (no VGPR round trip).
#pragma unroll
    for (int i = 0; i < 8; ++i) {
      const int e   = (tid + i * 256) * 8;   // half index into Vt
      const int d   = e >> 6;                // Vt row (64 halves per row)
      const int kkl = e & 63;
      const _Float16* src = VbT + (size_t)d * Lk + kb + kkl;
      const unsigned lds_off =
          (unsigned)(unsigned long long)&Vt[d][kkl];   // LDS byte address
      asm volatile("global_load_async_to_lds_b128 %0, %1, off"
                   :: "v"(lds_off), "v"(src)
                   : "memory");
    }
    // Prefetch next K chunk into cache while we compute.
    if (kb + kKC < Lk)
      __builtin_prefetch(Kb + (size_t)(kb + kKC) * kDim + lane * 64, 0, 1);

    // ---- S tile: Q(16x256) @ K^T(256x16); B-frag of K^T == A-frag of K.
    v8f sacc = {};
#pragma unroll
    for (int d = 0; d < 8; ++d) {
      v16h kf = load_a_frag_f16(Kb + (size_t)(kb + kt * 16) * kDim + d * 32,
                                kDim, lane);
      sacc = wmma_f16(qfrag[d], kf, sacc);
    }
    // Spill scaled scores (each wave writes its own 16x16 tile).
#pragma unroll
    for (int r = 0; r < 8; ++r) {
      const int rr = qh * 16 + r + 8 * (lane >> 4);
      const int cc = kt * 16 + (lane & 15);
      Ssc[rr][cc] = sacc[r] * scale;
    }
    asm volatile("s_wait_asynccnt 0x0" ::: "memory");  // Vt copies landed
    __syncthreads();                                   // Vt + Ssc complete

    // ---- online softmax: 8 threads per row (r = tid>>3, g = tid&7)
    const int r = tid >> 3;
    const int g = tid & 7;
    float pm = -3.0e38f;
#pragma unroll
    for (int c = 0; c < 8; ++c) pm = fmaxf(pm, Ssc[r][g * 8 + c]);
    red[r][g] = pm;
    __syncthreads();
    if (g == 0) {
      float cm = red[r][0];
#pragma unroll
      for (int c = 1; c < 8; ++c) cm = fmaxf(cm, red[r][c]);
      const float mn = fmaxf(mrow[r], cm);
      arow[r] = __expf(mrow[r] - mn);
      mrow[r] = mn;
    }
    __syncthreads();
    float psum = 0.0f;
    const float mn = mrow[r];
#pragma unroll
    for (int c = 0; c < 8; ++c) {
      const float p = __expf(Ssc[r][g * 8 + c] - mn);
      Ps[r][g * 8 + c] = (_Float16)p;
      psum += p;
    }
    red[r][g] = psum;
    __syncthreads();
    if (g == 0) {
      float s = 0.0f;
#pragma unroll
      for (int c = 0; c < 8; ++c) s += red[r][c];
      lrow[r] = lrow[r] * arow[r] + s;
    }
    __syncthreads();  // Ps complete; arow/lrow updated

    // ---- rescale O, then O += P(16x64) @ V(64x64 slice)
    float al[8];
#pragma unroll
    for (int rr = 0; rr < 8; ++rr) al[rr] = arow[qh * 16 + rr + 8 * (lane >> 4)];
#pragma unroll
    for (int t = 0; t < 4; ++t)
#pragma unroll
      for (int rr = 0; rr < 8; ++rr) oacc[t][rr] *= al[rr];

    v16h pf[2];
#pragma unroll
    for (int s = 0; s < 2; ++s)
      pf[s] = load_a_frag_f16(&Ps[qh * 16][0] + s * 32, kKC, lane);
#pragma unroll
    for (int t = 0; t < 4; ++t) {
#pragma unroll
      for (int s = 0; s < 2; ++s) {
        v16h vf = load_a_frag_f16(&Vt[kt * 64 + t * 16][0] + s * 32, kKC, lane);
        oacc[t] = wmma_f16(pf[s], vf, oacc[t]);
      }
    }
    __syncthreads();  // done with Vt/Ps/Ssc for this chunk
  }

  // ---- finalize: O /= l, store f16 x_attn
  float inv[8];
#pragma unroll
  for (int rr = 0; rr < 8; ++rr)
    inv[rr] = 1.0f / lrow[qh * 16 + rr + 8 * (lane >> 4)];
#pragma unroll
  for (int t = 0; t < 4; ++t) {
#pragma unroll
    for (int rr = 0; rr < 8; ++rr) {
      const int row = qbase + qh * 16 + rr + 8 * (lane >> 4);
      const int col = kt * 64 + t * 16 + (lane & 15);
      Xa[(size_t)b * Lq * kDim + (size_t)row * kDim + col] =
          (_Float16)(oacc[t][rr] * inv[rr]);
    }
  }
}

// ---------------------------------------------------------------------------
// Output projection + residual: out = Xa @ Wo + bo + x_dec   (fp32 out)
// ---------------------------------------------------------------------------
__global__ __launch_bounds__(128) void outproj_kernel(const _Float16* __restrict__ Xa,
                                                      const float* __restrict__ W,
                                                      const float* __restrict__ bias,
                                                      const float* __restrict__ resid,
                                                      float* __restrict__ Out) {
  __shared__ _Float16 Wt[64][32];

  const int tid  = threadIdx.x;
  const int lane = tid & 31;
  const int w    = tid >> 5;
  const int rbase = blockIdx.x * 16;
  const int cbase = blockIdx.y * 64;

  v8f acc = {};
  for (int kc = 0; kc < kDim; kc += 32) {
    for (int e = tid; e < 64 * 32; e += 128) {
      const int kk = e >> 6;
      const int n  = e & 63;
      Wt[n][kk] = (_Float16)W[(size_t)(kc + kk) * kDim + cbase + n];
    }
    __syncthreads();

    v16h a = load_a_frag_f16(Xa + (size_t)rbase * kDim + kc, kDim, lane);
    v16h b = load_a_frag_f16(&Wt[w * 16][0], 32, lane);
    acc = wmma_f16(a, b, acc);
    __syncthreads();
  }

  const int col = cbase + w * 16 + (lane & 15);
  const float bv = bias[col];
#pragma unroll
  for (int r = 0; r < 8; ++r) {
    const int row = rbase + r + 8 * (lane >> 4);
    Out[(size_t)row * kDim + col] = acc[r] + bv + resid[(size_t)row * kDim + col];
  }
}

// ---------------------------------------------------------------------------
extern "C" void kernel_launch(void* const* d_in, const int* in_sizes, int n_in,
                              void* d_out, int out_size, void* d_ws, size_t ws_size,
                              hipStream_t stream) {
  const float* x_dec = (const float*)d_in[0];
  const float* x_enc = (const float*)d_in[1];
  const float* Wq = (const float*)d_in[2];
  const float* bq = (const float*)d_in[3];
  const float* Wk = (const float*)d_in[4];
  const float* bk = (const float*)d_in[5];
  const float* Wv = (const float*)d_in[6];
  const float* bv = (const float*)d_in[7];
  const float* Wo = (const float*)d_in[8];
  const float* bo = (const float*)d_in[9];

  const int B = 4, L = kL;
  const int N = B * L;                       // 16384 rows
  const size_t mat = (size_t)N * kDim;       // elements per f16 intermediate

  _Float16* Qh  = (_Float16*)d_ws;           // 8 MB
  _Float16* Khw = Qh + mat;                  // 8 MB
  _Float16* VhT = Khw + mat;                 // 8 MB (transposed: [B][D][L])
  _Float16* Xa  = VhT + mat;                 // 8 MB

  dim3 pgrid(N / 16, kDim / 64);             // 1024 x 4 blocks, 128 threads
  proj_kernel <<<pgrid, 128, 0, stream>>>(x_dec, Wq, bq, Qh);
  proj_kernel <<<pgrid, 128, 0, stream>>>(x_enc, Wk, bk, Khw);
  projT_kernel<<<pgrid, 128, 0, stream>>>(x_enc, Wv, bv, VhT);

  dim3 agrid(L / 32, B);                     // 128 x 4 blocks, 256 threads
  attn_kernel<<<agrid, 256, 0, stream>>>(Qh, Khw, VhT, Xa, L, L);

  outproj_kernel<<<pgrid, 128, 0, stream>>>(Xa, Wo, bo, x_dec, (float*)d_out);
}
